// ClassAttention_85942295593249
// MI455X (gfx1250) — matrix operand order repaired and checked by reading, not compiled, not run on hardware
//
#include <hip/hip_runtime.h>
#include <hip/hip_bf16.h>
#include <cstdint>

#define B_SZ   64
#define N_TOK  1025
#define C_DIM  768
#define H_NUM  12
#define TM     128
#define KC     32
#define NKB    (C_DIM / KC)   // 24 K-steps
#define NT_TILES 9            // ceil(1025/128)

typedef __attribute__((ext_vector_type(16))) __bf16 v16bf;
typedef __attribute__((ext_vector_type(8)))  float  v8f;
typedef __attribute__((ext_vector_type(4)))  unsigned int u32x4;
typedef __attribute__((ext_vector_type(8)))  int          i32x8;
typedef __attribute__((ext_vector_type(4)))  int          i32x4;

union FragBF { v16bf v; uint4 q[2]; };

__device__ __forceinline__ unsigned short f2bf(float f) {
  unsigned u = __float_as_uint(f);
  u += 0x7FFFu + ((u >> 16) & 1u);           // round-to-nearest-even
  return (unsigned short)(u >> 16);
}
__device__ __forceinline__ float bf2f(unsigned short h) {
  return __uint_as_float(((unsigned)h) << 16);
}

#if __has_builtin(__builtin_amdgcn_tensor_load_to_lds)
#define HAVE_TDM 1
// DMA a [128 rows x 32 cols] bf16 tile (row stride C_DIM elems) from global
// into LDS with 16B padding after each 64B row (-> 80B LDS row stride).
__device__ __forceinline__ void tdm_load_tile(unsigned lds_addr, const unsigned short* gptr) {
  unsigned long long ga = (unsigned long long)(uintptr_t)gptr;
  u32x4 g0 = { 1u,                                   // count=1, is_restore=0
               lds_addr,                             // LDS byte address
               (unsigned)(ga & 0xFFFFFFFFu),         // global addr [31:0]
               (unsigned)((ga >> 32) & 0x01FFFFFFu) | (2u << 30) };  // [56:32] | type=2
  i32x8 g1 = { (int)((1u << 16)       // data_size = 1 -> 2 bytes
               | (1u << 20)           // pad_enable
               | (3u << 22)           // pad_interval: 2^(3+1)=16 DWORDs (64B)
               | (3u << 25)),         // pad_amount: 3+1 = 4 DWORDs (16B)
               (int)(((unsigned)C_DIM & 0xFFFFu) << 16),  // tensor_dim0 lo16 (768)
               (int)(((unsigned)C_DIM & 0xFFFFu) << 16),  // dim0 hi=0 | tensor_dim1 lo16 (768)
               (int)(32u << 16),                          // dim1 hi=0 | tile_dim0=32
               (int)128u,                                 // tile_dim1=128 | tile_dim2=0
               (int)C_DIM,                                // tensor_dim0_stride lo32 = 768
               0, 0 };                                    // stride hi / dim1_stride = 0
  i32x4 gz = { 0, 0, 0, 0 };
#if defined(__clang_major__) && __clang_major__ >= 23
  i32x8 gz8 = { 0, 0, 0, 0, 0, 0, 0, 0 };
  __builtin_amdgcn_tensor_load_to_lds(g0, g1, gz, gz, gz8, 0);
#else
  __builtin_amdgcn_tensor_load_to_lds(g0, g1, gz, gz, 0);
#endif
}
#endif

// ---------------------------------------------------------------- kernel 1
__global__ void cvt_weights(const float* __restrict__ kW, const float* __restrict__ vW,
                            unsigned short* __restrict__ kWb, unsigned short* __restrict__ vWb) {
  int idx = blockIdx.x * blockDim.x + threadIdx.x;
  if (idx < C_DIM * C_DIM) {
    kWb[idx] = f2bf(kW[idx]);
    vWb[idx] = f2bf(vW[idx]);
  }
}

// ---------------------------------------------------------------- kernel 2
// q = (cls @ qW^T + qb) * d^-0.5 ; cls = x[:, 0, :]
__global__ void q_proj(const float* __restrict__ x, const float* __restrict__ qW,
                       const float* __restrict__ qb, float* __restrict__ qs) {
  const int b = blockIdx.x;
  const int t = threadIdx.x;
  const float* xr = x + (size_t)b * N_TOK * C_DIM;   // token 0 row
  for (int j = 0; j < 3; ++j) {
    const int c = t + 256 * j;
    float acc = qb[c];
    const float* w = qW + (size_t)c * C_DIM;
    for (int i = 0; i < C_DIM; ++i) acc = fmaf(xr[i], w[i], acc);
    qs[(size_t)b * C_DIM + c] = acc * 0.125f;        // 64^-0.5
  }
}

// ---------------------------------------------------------------- kernel 3
// Fused: K/V projections (WMMA bf16, double-buffered TDM-staged weights),
// in-register q.k scores, talking-heads mix #1, logits out, V out (bf16).
__global__ __launch_bounds__(256) void kv_attn_scores(
    const float* __restrict__ x, const float* __restrict__ qs,
    const unsigned short* __restrict__ kWb, const unsigned short* __restrict__ vWb,
    const float* __restrict__ kb, const float* __restrict__ vb,
    const float* __restrict__ w1W, const float* __restrict__ w1b,
    float* __restrict__ logits, unsigned short* __restrict__ vout) {

  __shared__ unsigned short xt [TM][KC + 8];     // bf16 x tile, 80B row stride
  __shared__ unsigned short wkK[2][TM][KC + 8];  // double-buffered kW tiles
  __shared__ unsigned short wkV[2][TM][KC + 8];  // double-buffered vW tiles
  __shared__ float s_sc[TM][16];                 // raw scores [token][head]
  __shared__ float q_l[C_DIM];

  const int b    = blockIdx.y;
  const int n0   = blockIdx.x * TM;
  const int t    = threadIdx.x;
  const int lane = t & 31;
  const int wv   = t >> 5;
  const int rb   = wv * 16;                   // this wave's 16-row block
  const int lhalf = lane >> 4;                // 0 | 1
  const int l15   = lane & 15;

  for (int c = t; c < C_DIM; c += 256) q_l[c] = qs[(size_t)b * C_DIM + c];

  const float* xbase = x + (size_t)b * N_TOK * C_DIM;
  const int xrow  = t >> 3;          // x-staging: row slice this thread covers
  const int xcol4 = (t & 7) * 4;
  const v8f vzero = {0.f, 0.f, 0.f, 0.f, 0.f, 0.f, 0.f, 0.f};

  for (int cc = 0; cc < 6; ++cc) {            // 6 chunks of 128 output cols (= 2 heads)
    v8f acc_k[8], acc_v[8];
    #pragma unroll
    for (int i = 0; i < 8; ++i) { acc_k[i] = vzero; acc_v[i] = vzero; }

    const unsigned short* kWcc = kWb + (size_t)(cc * 128) * C_DIM;
    const unsigned short* vWcc = vWb + (size_t)(cc * 128) * C_DIM;

    // ---------------- prologue: stage K-step 0 into buffer 0
#ifdef HAVE_TDM
    if (wv == 0) {
      tdm_load_tile((unsigned)(uintptr_t)&wkK[0][0][0], kWcc);
      tdm_load_tile((unsigned)(uintptr_t)&wkV[0][0][0], vWcc);
    }
#else
    #pragma unroll
    for (int mtx = 0; mtx < 2; ++mtx) {
      const unsigned short* src = mtx ? vWcc : kWcc;
      for (int c = t; c < 512; c += 256) {
        const int row = c >> 2, part = c & 3;
        const uint4 d = *reinterpret_cast<const uint4*>(src + (size_t)row * C_DIM + part * 8);
        *reinterpret_cast<uint4*>(mtx ? &wkV[0][row][part * 8] : &wkK[0][row][part * 8]) = d;
      }
    }
#endif
    #pragma unroll
    for (int p = 0; p < 4; ++p) {           // stage x tile for K-step 0
      int row = p * 32 + xrow;
      int n = n0 + row; if (n > N_TOK - 1) n = N_TOK - 1;
      const float4 f = *reinterpret_cast<const float4*>(xbase + (size_t)n * C_DIM + xcol4);
      unsigned lo = (unsigned)f2bf(f.x) | ((unsigned)f2bf(f.y) << 16);
      unsigned hi = (unsigned)f2bf(f.z) | ((unsigned)f2bf(f.w) << 16);
      uint2 u; u.x = lo; u.y = hi;
      *reinterpret_cast<uint2*>(&xt[row][xcol4]) = u;
    }
#ifdef HAVE_TDM
    if (wv == 0) __builtin_amdgcn_s_wait_tensorcnt(0);
#endif
    __syncthreads();

    // ---------------- main K loop, weights double-buffered
    for (int kbk = 0; kbk < NKB; ++kbk) {
      const int  cur  = kbk & 1;
      const int  nx   = cur ^ 1;
      const bool more = (kbk + 1 < NKB);
      const int  kc0n = (kbk + 1) * KC;

#ifdef HAVE_TDM
      if (more && wv == 0) {   // kick next weight DMAs; they overlap the WMMAs below
        tdm_load_tile((unsigned)(uintptr_t)&wkK[nx][0][0], kWcc + kc0n);
        tdm_load_tile((unsigned)(uintptr_t)&wkV[nx][0][0], vWcc + kc0n);
      }
#endif
      // A fragment for this K-step (into registers)
      FragBF a;
      {
        const int m  = rb + l15;
        const int lo = lhalf ? 16 : 0;
        a.q[0] = *reinterpret_cast<const uint4*>(&xt[m][lo]);
        a.q[1] = *reinterpret_cast<const uint4*>(&xt[m][lo + 8]);
      }
      // prefetch next x tile into registers (latency hidden under WMMA)
      float4 xf[4];
      if (more) {
        #pragma unroll
        for (int p = 0; p < 4; ++p) {
          int row = p * 32 + xrow;
          int n = n0 + row; if (n > N_TOK - 1) n = N_TOK - 1;
          const float* src = xbase + (size_t)n * C_DIM + kc0n + xcol4;
          xf[p] = *reinterpret_cast<const float4*>(src);
          if (kbk + 2 < NKB) __builtin_prefetch(src + KC, 0, 0);
        }
      }
      __syncthreads();   // all A fragments secured -> xt may be overwritten

      // 16 WMMAs with 2-deep register pipeline on the B fragments
      const int kk = lhalf ? 16 : 0;
      FragBF bkf[2], bvf[2];
      {
        const int r = l15;
        bkf[0].q[0] = *reinterpret_cast<const uint4*>(&wkK[cur][r][kk]);
        bkf[0].q[1] = *reinterpret_cast<const uint4*>(&wkK[cur][r][kk + 8]);
        bvf[0].q[0] = *reinterpret_cast<const uint4*>(&wkV[cur][r][kk]);
        bvf[0].q[1] = *reinterpret_cast<const uint4*>(&wkV[cur][r][kk + 8]);
      }
      #pragma unroll
      for (int t8 = 0; t8 < 8; ++t8) {
        const int pb = t8 & 1, pn = pb ^ 1;
        if (t8 < 7) {
          const int r = (t8 + 1) * 16 + l15;
          bkf[pn].q[0] = *reinterpret_cast<const uint4*>(&wkK[cur][r][kk]);
          bkf[pn].q[1] = *reinterpret_cast<const uint4*>(&wkK[cur][r][kk + 8]);
          bvf[pn].q[0] = *reinterpret_cast<const uint4*>(&wkV[cur][r][kk]);
          bvf[pn].q[1] = *reinterpret_cast<const uint4*>(&wkV[cur][r][kk + 8]);
        }
        acc_k[t8] = __builtin_amdgcn_wmma_f32_16x16x32_bf16(
            false, a.v, false, bkf[pb].v, (short)0, acc_k[t8], false, false);
        acc_v[t8] = __builtin_amdgcn_wmma_f32_16x16x32_bf16(
            false, a.v, false, bvf[pb].v, (short)0, acc_v[t8], false, false);
      }

      if (more) {   // write next x tile (bf16) into xt
        #pragma unroll
        for (int p = 0; p < 4; ++p) {
          int row = p * 32 + xrow;
          unsigned lo = (unsigned)f2bf(xf[p].x) | ((unsigned)f2bf(xf[p].y) << 16);
          unsigned hi = (unsigned)f2bf(xf[p].z) | ((unsigned)f2bf(xf[p].w) << 16);
          uint2 u; u.x = lo; u.y = hi;
          *reinterpret_cast<uint2*>(&xt[row][xcol4]) = u;
        }
      }
#ifndef HAVE_TDM
      if (more) {   // cooperative staging of next weight tiles
        #pragma unroll
        for (int mtx = 0; mtx < 2; ++mtx) {
          const unsigned short* src = (mtx ? vWcc : kWcc) + kc0n;
          for (int c = t; c < 512; c += 256) {
            const int row = c >> 2, part = c & 3;
            const uint4 d = *reinterpret_cast<const uint4*>(src + (size_t)row * C_DIM + part * 8);
            *reinterpret_cast<uint4*>(mtx ? &wkV[nx][row][part * 8]
                                          : &wkK[nx][row][part * 8]) = d;
          }
        }
      }
#else
      if (more && wv == 0) __builtin_amdgcn_s_wait_tensorcnt(0);
#endif
      __syncthreads();
    }

    // scores: s[m,h] = sum_d q[h*64+d] * (k[m,h*64+d]+kb);  V -> bf16 store
    float part[2][8];
    #pragma unroll
    for (int h = 0; h < 2; ++h)
      #pragma unroll
      for (int i = 0; i < 8; ++i) part[h][i] = 0.f;

    #pragma unroll
    for (int t8 = 0; t8 < 8; ++t8) {
      const int   c_l = cc * 128 + t8 * 16 + l15;
      const float qv  = q_l[c_l];
      const float kbv = kb[c_l];
      const float vbv = vb[c_l];
      const int   hh  = t8 >> 2;                      // which of the 2 heads in chunk
      #pragma unroll
      for (int i = 0; i < 8; ++i)
        part[hh][i] = fmaf(qv, acc_k[t8][i] + kbv, part[hh][i]);
      #pragma unroll
      for (int i = 0; i < 8; ++i) {
        const int m = rb + i + lhalf * 8;
        const int n = n0 + m;
        if (n < N_TOK)
          vout[((size_t)b * N_TOK + n) * C_DIM + c_l] = f2bf(acc_v[t8][i] + vbv);
      }
    }
    #pragma unroll
    for (int hh = 0; hh < 2; ++hh)
      #pragma unroll
      for (int i = 0; i < 8; ++i) {
        float v = part[hh][i];
        v += __shfl_xor(v, 1);
        v += __shfl_xor(v, 2);
        v += __shfl_xor(v, 4);
        v += __shfl_xor(v, 8);          // reduce 16 columns within each half-wave
        if (l15 == 0) {
          const int m = rb + i + lhalf * 8;
          s_sc[m][cc * 2 + hh] = v;
        }
      }
  }
  __syncthreads();

  // talking-heads mix #1: logit[g] = s[g] + sum_h s[h]*w1W[g,h] + w1b[g]
  for (int idx = t; idx < TM * H_NUM; idx += 256) {
    const int m = idx / H_NUM;
    const int g = idx - m * H_NUM;
    const int n = n0 + m;
    if (n < N_TOK) {
      float acc = s_sc[m][g] + w1b[g];
      #pragma unroll
      for (int h = 0; h < H_NUM; ++h) acc = fmaf(s_sc[m][h], w1W[g * H_NUM + h], acc);
      logits[((size_t)b * H_NUM + g) * N_TOK + n] = acc;
    }
  }
}

// ---------------------------------------------------------------- kernel 4
// Per batch: softmax over keys, mix #2 on probabilities, attn @ V
__global__ __launch_bounds__(256) void softmax_mix_av(
    const float* __restrict__ logits, const unsigned short* __restrict__ vws,
    const float* __restrict__ w2W, const float* __restrict__ w2b,
    float* __restrict__ attout) {

  __shared__ float p[H_NUM][N_TOK + 3];   // ~49.3 KB
  __shared__ float redm[8][H_NUM];
  __shared__ float reds[8][H_NUM];
  __shared__ float mxs[H_NUM];
  __shared__ float zs[H_NUM];

  const int b    = blockIdx.x;
  const int t    = threadIdx.x;
  const int lane = t & 31;
  const int wv   = t >> 5;
  const float* lg = logits + (size_t)b * H_NUM * N_TOK;

  for (int h = 0; h < H_NUM; ++h) {
    const float* lh = lg + (size_t)h * N_TOK;
    float m = -3.4e38f;
    for (int k = t; k < N_TOK; k += 256) { float v = lh[k]; p[h][k] = v; m = fmaxf(m, v); }
    #pragma unroll
    for (int off = 16; off >= 1; off >>= 1) m = fmaxf(m, __shfl_xor(m, off));
    if (lane == 0) redm[wv][h] = m;
  }
  __syncthreads();
  if (t < H_NUM) {
    float m = redm[0][t];
    for (int w = 1; w < 8; ++w) m = fmaxf(m, redm[w][t]);
    mxs[t] = m;
  }
  __syncthreads();
  for (int h = 0; h < H_NUM; ++h) {
    const float mh = mxs[h];
    float s = 0.f;
    for (int k = t; k < N_TOK; k += 256) { float e = __expf(p[h][k] - mh); p[h][k] = e; s += e; }
    #pragma unroll
    for (int off = 16; off >= 1; off >>= 1) s += __shfl_xor(s, off);
    if (lane == 0) reds[wv][h] = s;
  }
  __syncthreads();
  if (t < H_NUM) {
    float z = 0.f;
    for (int w = 0; w < 8; ++w) z += reds[w][t];
    zs[t] = z;
  }
  __syncthreads();

  // normalize + mix #2, column-wise in place
  for (int k = t; k < N_TOK; k += 256) {
    float pn[H_NUM], a2[H_NUM];
    #pragma unroll
    for (int h = 0; h < H_NUM; ++h) pn[h] = p[h][k] / zs[h];
    #pragma unroll
    for (int g = 0; g < H_NUM; ++g) {
      float a = pn[g] + w2b[g];
      #pragma unroll
      for (int h = 0; h < H_NUM; ++h) a = fmaf(pn[h], w2W[g * H_NUM + h], a);
      a2[g] = a;
    }
    #pragma unroll
    for (int g = 0; g < H_NUM; ++g) p[g][k] = a2[g];
  }
  __syncthreads();

  // out[c] = sum_k attn2[g][k] * v[k][c],  g = c/64
  #pragma unroll
  for (int j = 0; j < 3; ++j) {
    const int c = t + 256 * j;
    const int g = c >> 6;
    const unsigned short* vp = vws + (size_t)b * N_TOK * C_DIM + c;
    float acc = 0.f;
    for (int k = 0; k < N_TOK; ++k)
      acc = fmaf(p[g][k], bf2f(vp[(size_t)k * C_DIM]), acc);
    attout[(size_t)b * C_DIM + c] = acc;
  }
}

// ---------------------------------------------------------------- kernel 5
__global__ void out_proj(const float* __restrict__ attout, const float* __restrict__ projW,
                         const float* __restrict__ projb, float* __restrict__ y) {
  const int b = blockIdx.x;
  const int t = threadIdx.x;
  const float* a = attout + (size_t)b * C_DIM;
  for (int j = 0; j < 3; ++j) {
    const int c = t + 256 * j;
    float acc = projb[c];
    const float* w = projW + (size_t)c * C_DIM;
    for (int i = 0; i < C_DIM; ++i) acc = fmaf(a[i], w[i], acc);
    y[(size_t)b * C_DIM + c] = acc;
  }
}

// ----------------------------------------------------------------
extern "C" void kernel_launch(void* const* d_in, const int* in_sizes, int n_in,
                              void* d_out, int out_size, void* d_ws, size_t ws_size,
                              hipStream_t stream) {
  const float* x   = (const float*)d_in[0];
  const float* qW  = (const float*)d_in[1];
  const float* qb  = (const float*)d_in[2];
  const float* kW  = (const float*)d_in[3];
  const float* kb  = (const float*)d_in[4];
  const float* vW  = (const float*)d_in[5];
  const float* vb  = (const float*)d_in[6];
  const float* w1W = (const float*)d_in[7];
  const float* w1b = (const float*)d_in[8];
  const float* w2W = (const float*)d_in[9];
  const float* w2b = (const float*)d_in[10];
  const float* pW  = (const float*)d_in[11];
  const float* pb  = (const float*)d_in[12];

  char* ws = (char*)d_ws;
  size_t off = 0;
  auto take = [&](size_t bytes) -> void* {
    void* p = ws + off;
    off = (off + bytes + 255) & ~(size_t)255;
    return p;
  };
  float*          qs     = (float*)take((size_t)B_SZ * C_DIM * 4);
  unsigned short* kWb    = (unsigned short*)take((size_t)C_DIM * C_DIM * 2);
  unsigned short* vWb    = (unsigned short*)take((size_t)C_DIM * C_DIM * 2);
  float*          logits = (float*)take((size_t)B_SZ * H_NUM * N_TOK * 4);
  float*          attout = (float*)take((size_t)B_SZ * C_DIM * 4);
  unsigned short* vws    = (unsigned short*)take((size_t)B_SZ * N_TOK * C_DIM * 2);

  cvt_weights<<<dim3((C_DIM * C_DIM + 255) / 256), dim3(256), 0, stream>>>(kW, vW, kWb, vWb);
  q_proj<<<dim3(B_SZ), dim3(256), 0, stream>>>(x, qW, qb, qs);
  kv_attn_scores<<<dim3(NT_TILES, B_SZ), dim3(256), 0, stream>>>(
      x, qs, kWb, vWb, kb, vb, w1W, w1b, logits, vws);
  softmax_mix_av<<<dim3(B_SZ), dim3(256), 0, stream>>>(logits, vws, w2W, w2b, attout);
  out_proj<<<dim3(B_SZ), dim3(256), 0, stream>>>(attout, pW, pb, (float*)d_out);
}
